// SBC_75419625718484
// MI455X (gfx1250) — matrix operand order repaired and checked by reading, not compiled, not run on hardware
//
#include <hip/hip_runtime.h>
#include <hip/hip_bf16.h>
#include <cstdint>

typedef __attribute__((ext_vector_type(16))) _Float16 v16h;
typedef __attribute__((ext_vector_type(8)))  _Float16 v8h;
typedef __attribute__((ext_vector_type(4)))  _Float16 v4h;
typedef __attribute__((ext_vector_type(8)))  float    v8f;

// ---------------------------------------------------------------------------
// Problem constants
// ---------------------------------------------------------------------------
#define BS    256
#define C0    256       // input channels
#define HW    256       // 16*16 pixels
#define CCAT  1024      // concat channels = 2*(C+HW)
#define KCONV 9216      // 1024 * 9 taps
#define CORR_ELEMS (BS * HW * HW)   // 16,777,216 per correlation

#define CHUNK    64     // ic per LDS staging chunk
#define LDS_ROWH 72     // halfs per pixel row in LDS (144B: 16B-aligned, conflict-free)

#define USE_ASYNC 1     // global_load_async_to_lds_b128 staging path

// ---------------------------------------------------------------------------
// WMMA fragment helpers (wave32; A and B both use contiguous k = 16*(lane>>4) + e)
// ---------------------------------------------------------------------------

// A-matrix 16x32 f16 from row-major [M][lda] at (m0,k0); lda, k0 16B-aligned.
// lane 0-15: M=lane, K k0+0..15; lane 16-31: K k0+16..31 (contiguous)
static __device__ inline v16h load_a_frag(const _Float16* __restrict__ A, int lda,
                                          int m0, int k0, int lane) {
    int row = m0 + (lane & 15);
    int kb  = k0 + ((lane >> 4) ? 16 : 0);
    const _Float16* p = A + (size_t)row * lda + kb;
    v8h lo = *(const v8h*)p;          // K = kb .. kb+7
    v8h hi = *(const v8h*)(p + 8);    // K = kb+8 .. kb+15
    return __builtin_shufflevector(lo, hi, 0,1,2,3,4,5,6,7,8,9,10,11,12,13,14,15);
}

// B-matrix 32x16 f16 from a TRANSPOSED row-major source BT[N][ldb=K] at (k0,n0).
// lane: column N = n0+(lane&15); elements e: K = k0 + 16*(lane>>4) + e (contiguous).
static __device__ inline v16h load_bt_frag(const _Float16* __restrict__ BT, int ldb,
                                           int k0, int n0, int lane) {
    int col = n0 + (lane & 15);
    int kb  = k0 + ((lane >> 4) ? 16 : 0);
    const _Float16* p = BT + (size_t)col * ldb + kb;
    v8h lo = *(const v8h*)p;
    v8h hi = *(const v8h*)(p + 8);
    return __builtin_shufflevector(lo, hi, 0,1,2,3,4,5,6,7,8,9,10,11,12,13,14,15);
}

// Conv B fragment from LDS chunk xt[px][LDS_ROWH] (ic-contiguous rows).
// Zero-padded for out-of-image shifted pixels.
static __device__ inline v16h load_b_lds(const _Float16* xt, int tap, int icl,
                                         int n0, int lane) {
    int n = n0 + (lane & 15);
    int y = (n >> 4) + (tap / 3) - 1;
    int x = (n & 15) + (tap % 3) - 1;
    v16h r = {};
    if ((unsigned)y < 16u && (unsigned)x < 16u) {
        const _Float16* p = xt + (y * 16 + x) * LDS_ROWH + icl + ((lane >> 4) ? 16 : 0);
        v8h lo = *(const v8h*)p;
        v8h hi = *(const v8h*)(p + 8);
        r = __builtin_shufflevector(lo, hi, 0,1,2,3,4,5,6,7,8,9,10,11,12,13,14,15);
    }
    return r;
}

// ---------------------------------------------------------------------------
// Kernel 1: ReLU + L2-normalize over channels.
// Outputs: fnt [b][px][c] f16 (WMMA A & B operand for correlation) and the
// ReLU'd channels of the TRANSPOSED concat buffer featcatT [b][px][1024].
// grid = BS, block = 256 (thread = pixel) -> all stores contiguous.
// ---------------------------------------------------------------------------
__global__ void relu_norm_kernel(const float* __restrict__ f,
                                 _Float16* __restrict__ fnt,      // [b][px][c]
                                 _Float16* __restrict__ featcatT, // [b][px][1024]
                                 int choff) {
    int b = blockIdx.x, t = threadIdx.x;
    const float* fb = f + (size_t)b * C0 * HW + t;
    float ss = 0.f;
    for (int c = 0; c < C0; ++c) {
        float v = fmaxf(fb[(size_t)c * HW], 0.f);
        ss += v * v;
    }
    float inv = 1.0f / fmaxf(sqrtf(ss), 1e-12f);
    _Float16* fcb  = featcatT + ((size_t)b * HW + t) * CCAT + choff;
    _Float16* fntb = fnt + (size_t)b * C0 * HW + (size_t)t * C0;
    for (int c = 0; c < C0; c += 4) {
        v4h hv, nv;
#pragma unroll
        for (int j = 0; j < 4; ++j) {
            float v = fmaxf(fb[(size_t)(c + j) * HW], 0.f);
            hv[j] = (_Float16)v;
            nv[j] = (_Float16)(v * inv);
        }
        *(v4h*)(fcb + c)  = hv;   // contiguous, 8B aligned
        *(v4h*)(fntb + c) = nv;   // contiguous, 8B aligned
    }
}

// ---------------------------------------------------------------------------
// Kernel 2: batched correlation GEMM, both operands from transposed buffers
// (all fragment loads contiguous b128). Wave tile 16x64; grid (8, BS, 2).
// fp32 result -> d_out (reference [k][px] layout); f16 -> featcatT (contiguous).
// ---------------------------------------------------------------------------
__global__ void corr_wmma_kernel(const _Float16* __restrict__ f1nt,
                                 const _Float16* __restrict__ f2nt,
                                 float* __restrict__ corr_out,    // d_out + 512
                                 _Float16* __restrict__ featcatT) {
    int b     = blockIdx.y;
    int which = blockIdx.z;                 // 0 = corr1, 1 = corr2
    const _Float16* A  = (which == 0 ? f2nt : f1nt) + (size_t)b * C0 * HW;
    const _Float16* BT = (which == 0 ? f1nt : f2nt) + (size_t)b * C0 * HW;
    float* Cout = corr_out + (size_t)which * CORR_ELEMS + (size_t)b * HW * HW;
    int chbase  = 256 + which * 512;

    int lane = threadIdx.x & 31;
    int wid  = threadIdx.x >> 5;
    int wt   = blockIdx.x * 8 + wid;        // 0..63
    int m0   = (wt >> 2) * 16;
    int n0   = (wt & 3) * 64;

    v8f acc[4] = {};
    for (int kk = 0; kk < C0; kk += 32) {
        v16h a = load_a_frag(A, C0, m0, kk, lane);
#pragma unroll
        for (int j = 0; j < 4; ++j) {
            v16h bb = load_bt_frag(BT, C0, kk, n0 + 16 * j, lane);
            acc[j] = __builtin_amdgcn_wmma_f32_16x16x32_f16(
                false, a, false, bb, (short)0, acc[j], false, false);
        }
    }
    int col = lane & 15;
    int rb  = (lane >> 4) ? 8 : 0;
#pragma unroll
    for (int j = 0; j < 4; ++j) {
        int nb = n0 + 16 * j + col;
        v8h h;
#pragma unroll
        for (int r = 0; r < 8; ++r) {
            float v = acc[j][r];
            Cout[(size_t)(m0 + rb + r) * HW + nb] = v;   // reference layout (strided)
            h[r] = (_Float16)v;
        }
        // transposed concat: channels m0+rb..+7 contiguous at pixel nb (16B aligned)
        *(v8h*)(featcatT + ((size_t)b * HW + nb) * CCAT + chbase + m0 + rb) = h;
    }
}

// ---------------------------------------------------------------------------
// Kernel 3: repack conv weights [oc][ic][3][3] f32 -> [oc][tap][ic] f16
// ---------------------------------------------------------------------------
__global__ void repack_w_kernel(const float* __restrict__ w,
                                _Float16* __restrict__ wp, int OC) {
    size_t i = (size_t)blockIdx.x * 256 + threadIdx.x;
    size_t total = (size_t)OC * KCONV;
    if (i >= total) return;
    int oc  = (int)(i / KCONV);
    int rem = (int)(i % KCONV);
    int tap = rem >> 10;
    int ic  = rem & 1023;
    wp[i] = (_Float16)w[(size_t)oc * KCONV + (size_t)ic * 9 + tap];
}

// ---------------------------------------------------------------------------
// Kernel 4: 3x3 SAME conv as implicit WMMA GEMM.
// Input XT [b][px][1024] (transposed) -> LDS staging is a straight b128 copy,
// done with gfx1250 async-to-LDS (ASYNCcnt) when USE_ASYNC.
// Wave tile 32x64: per k-step 2 A-fragments (4 global b128) + 4 shared
// B-fragments (8 ds b128) feed 8 WMMAs. Output YT [b][px][M], contiguous v8h
// stores, fused bias + ReLU. grid = (M/64, BS), block = 256 (8 waves).
// ---------------------------------------------------------------------------
__global__ void conv3x3_wmma_kernel(const _Float16* __restrict__ XT,  // [b][256][1024]
                                    const _Float16* __restrict__ W,   // [M][9216]
                                    const float* __restrict__ bias,   // [M]
                                    _Float16* __restrict__ YT,        // [b][256][M]
                                    int M) {
    __shared__ _Float16 xt[HW * LDS_ROWH];   // 36,864 B

    int b = blockIdx.y;
    const _Float16* XTb = XT + (size_t)b * HW * CCAT;
    int lane  = threadIdx.x & 31;
    int wid   = threadIdx.x >> 5;
    int mhalf = wid >> 2;                   // 0..1
    int m0    = blockIdx.x * 64 + mhalf * 32;
    int n0    = (wid & 3) * 64;

    v8f acc0[4] = {};
    v8f acc1[4] = {};
    for (int c0 = 0; c0 < CCAT; c0 += CHUNK) {
        __syncthreads();   // previous chunk fully consumed before overwrite
        // ---- stage CHUNK channels x 256 px: straight 16B copies ----
        // 2048 b128 segments, 8 per thread.
#pragma unroll
        for (int i = 0; i < 8; ++i) {
            int s   = i * 256 + threadIdx.x;   // 0..2047
            int px  = s >> 3;
            int seg = s & 7;
            const char* g = (const char*)(XTb + (size_t)px * CCAT + c0) + seg * 16;
#if USE_ASYNC
            uint32_t ldsoff = (uint32_t)(uintptr_t)xt + (uint32_t)(px * (LDS_ROWH * 2) + seg * 16);
            asm volatile("global_load_async_to_lds_b128 %0, %1, off"
                         :: "v"(ldsoff), "v"((uint64_t)(uintptr_t)g)
                         : "memory");
#else
            *(v8h*)((char*)xt + px * (LDS_ROWH * 2) + seg * 16) = *(const v8h*)g;
#endif
        }
#if USE_ASYNC
        asm volatile("s_wait_asynccnt 0x0" ::: "memory");
#endif
        __syncthreads();
        // ---- compute: 9 taps x (CHUNK/32) ksteps, 8 WMMA each ----
        for (int tap = 0; tap < 9; ++tap) {
#pragma unroll
            for (int ks = 0; ks < CHUNK / 32; ++ks) {
                int icl = ks * 32;
                int kg  = tap * 1024 + c0 + icl;
                v16h a0 = load_a_frag(W, KCONV, m0,      kg, lane);
                v16h a1 = load_a_frag(W, KCONV, m0 + 16, kg, lane);
#pragma unroll
                for (int j = 0; j < 4; ++j) {
                    v16h bb = load_b_lds(xt, tap, icl, n0 + 16 * j, lane);
                    acc0[j] = __builtin_amdgcn_wmma_f32_16x16x32_f16(
                        false, a0, false, bb, (short)0, acc0[j], false, false);
                    acc1[j] = __builtin_amdgcn_wmma_f32_16x16x32_f16(
                        false, a1, false, bb, (short)0, acc1[j], false, false);
                }
            }
        }
    }
    // ---- epilogue: bias + relu, contiguous v8h stores into YT ----
    int col = lane & 15;
    int rb  = (lane >> 4) ? 8 : 0;
    _Float16* YTb = YT + (size_t)b * HW * M;
#pragma unroll
    for (int j = 0; j < 4; ++j) {
        int nb = n0 + 16 * j + col;
        v8h h0, h1;
#pragma unroll
        for (int r = 0; r < 8; ++r) {
            h0[r] = (_Float16)fmaxf(acc0[j][r] + bias[m0 + rb + r], 0.f);
            h1[r] = (_Float16)fmaxf(acc1[j][r] + bias[m0 + 16 + rb + r], 0.f);
        }
        *(v8h*)(YTb + (size_t)nb * M + m0 + rb)      = h0;
        *(v8h*)(YTb + (size_t)nb * M + m0 + 16 + rb) = h1;
    }
}

// ---------------------------------------------------------------------------
// Kernel 5: cls head. PSROI diagonal means only o = c*256 + pixel matters:
// cls_score[b][c] = mean_px relu( dot(y2T[b][px][:], Wcls[c*256+px][:]) )
// then softmax over 2 classes. grid = BS, block = 256 (thread = pixel).
// y2T is [b][px][512] -> fully contiguous dot products.
// ---------------------------------------------------------------------------
__global__ void cls_head_kernel(const _Float16* __restrict__ y2T,  // [b][256][512]
                                const float* __restrict__ wcls,    // [512][512]
                                float* __restrict__ out) {         // [b][2]
    __shared__ float red[2][256];
    int b = blockIdx.x, t = threadIdx.x;
    const _Float16* yb = y2T + ((size_t)b * HW + t) * 512;
#pragma unroll
    for (int c = 0; c < 2; ++c) {
        const float* w = wcls + (size_t)(c * 256 + t) * 512;
        float s = 0.f;
        for (int ch = 0; ch < 512; ++ch)
            s += (float)yb[ch] * w[ch];
        red[c][t] = fmaxf(s, 0.f);
    }
    __syncthreads();
    for (int off = 128; off > 0; off >>= 1) {
        if (t < off) {
            red[0][t] += red[0][t + off];
            red[1][t] += red[1][t + off];
        }
        __syncthreads();
    }
    if (t == 0) {
        float s0 = red[0][0] * (1.0f / 256.0f);
        float s1 = red[1][0] * (1.0f / 256.0f);
        float m  = fmaxf(s0, s1);
        float e0 = expf(s0 - m), e1 = expf(s1 - m);
        float inv = 1.0f / (e0 + e1);
        out[(size_t)b * 2 + 0] = e0 * inv;
        out[(size_t)b * 2 + 1] = e1 * inv;
    }
}

// ---------------------------------------------------------------------------
// Launcher
// ---------------------------------------------------------------------------
extern "C" void kernel_launch(void* const* d_in, const int* in_sizes, int n_in,
                              void* d_out, int out_size, void* d_ws, size_t ws_size,
                              hipStream_t stream) {
    const float* feature_1  = (const float*)d_in[0];
    const float* feature_2  = (const float*)d_in[1];
    const float* conv1_w    = (const float*)d_in[2];
    const float* conv1_b    = (const float*)d_in[3];
    const float* conv2_w    = (const float*)d_in[4];
    const float* conv2_b    = (const float*)d_in[5];
    const float* conv_cls_w = (const float*)d_in[6];
    float* out = (float*)d_out;

    // workspace carve-up (bytes)
    char* ws = (char*)d_ws;
    const size_t SZ_NORM = (size_t)BS * C0 * HW * 2;      //  33,554,432
    const size_t SZ_CAT  = (size_t)BS * CCAT * HW * 2;    // 134,217,728
    const size_t SZ_Y2   = (size_t)BS * 512 * HW * 2;     //  67,108,864
    const size_t SZ_W1   = (size_t)1024 * KCONV * 2;      //  18,874,368
    size_t off = 0;
    _Float16* f1nt     = (_Float16*)(ws + off); off += SZ_NORM;  // [b][px][256]
    _Float16* f2nt     = (_Float16*)(ws + off); off += SZ_NORM;  // [b][px][256]
    _Float16* featcatT = (_Float16*)(ws + off); off += SZ_CAT;   // [b][px][1024]
    _Float16* y1T      = (_Float16*)(ws + off); off += SZ_CAT;   // [b][px][1024]
    _Float16* y2T      = (_Float16*)(ws + off); off += SZ_Y2;    // [b][px][512]
    _Float16* w1p      = (_Float16*)(ws + off); off += SZ_W1;
    _Float16* w2p      = (_Float16*)(ws + off);

    // weight repack (f32 -> f16, [oc][tap][ic])
    repack_w_kernel<<<(1024u * KCONV + 255) / 256, 256, 0, stream>>>(conv1_w, w1p, 1024);
    repack_w_kernel<<<(512u  * KCONV + 255) / 256, 256, 0, stream>>>(conv2_w, w2p, 512);

    // relu + normalize (also fills concat channels 0-255 / 512-767)
    relu_norm_kernel<<<BS, 256, 0, stream>>>(feature_1, f1nt, featcatT, 0);
    relu_norm_kernel<<<BS, 256, 0, stream>>>(feature_2, f2nt, featcatT, 512);

    // correlations: fp32 to d_out[512..], f16 into concat channels 256-511 / 768-1023
    corr_wmma_kernel<<<dim3(8, BS, 2), 256, 0, stream>>>(
        f1nt, f2nt, out + 512, featcatT);

    // conv1: 1024 -> 1024, fused bias+relu
    conv3x3_wmma_kernel<<<dim3(1024 / 64, BS), 256, 0, stream>>>(
        featcatT, w1p, conv1_b, y1T, 1024);

    // conv2: 1024 -> 512, fused bias+relu
    conv3x3_wmma_kernel<<<dim3(512 / 64, BS), 256, 0, stream>>>(
        y1T, w2p, conv2_b, y2T, 512);

    // cls head: diagonal 1x1 conv + relu + avgpool + softmax -> d_out[0..511]
    cls_head_kernel<<<BS, 256, 0, stream>>>(y2T, conv_cls_w, out);
}
